// FvdbUpBlock_8804682957042
// MI455X (gfx1250) — compile-verified, matrix operand order checked by reading, not run on hardware
//
#include <hip/hip_runtime.h>
#include <hip/hip_bf16.h>
#include <stdint.h>

typedef __attribute__((ext_vector_type(16))) _Float16 v16h;
typedef __attribute__((ext_vector_type(8)))  _Float16 v8h;
typedef __attribute__((ext_vector_type(8)))  float    v8f;
typedef __attribute__((ext_vector_type(4)))  float    v4f;

#define SLOPE 0.01f
#define EPS   1e-5f

// ---------------------------------------------------------------------------
// WMMA fragment helpers (16x16x32 f16, wave32).
// A 16x32 f16 layout: lane m = lane&15 holds row m; g = lane>>4 selects K half.
//   element e (0..15) -> K = 32*kchunk + 8*g + (e & 7) + 16*(e >> 3)
// i.e. two contiguous 8-half (16B) chunks at +8g and +16+8g within the k-chunk.
// B uses the same lane pattern with "row" = output column (we stage W
// transposed [N][K] in LDS so B fragments are the same two 16B loads).
// ---------------------------------------------------------------------------
__device__ __forceinline__ v16h frag_combine(v8h lo, v8h hi) {
  v16h r;
#pragma unroll
  for (int i = 0; i < 8; ++i) { r[i] = lo[i]; r[i + 8] = hi[i]; }
  return r;
}

__device__ __forceinline__ v16h load_wt_frag(const _Float16* wt_row, int k0, int g) {
  v8h lo = *(const v8h*)(wt_row + k0 + 8 * g);
  v8h hi = *(const v8h*)(wt_row + k0 + 16 + 8 * g);
  return frag_combine(lo, hi);
}

// fp32 A chunk (2x 16B NT vector loads per half) -> f16 fragment
__device__ __forceinline__ v16h load_a_f32_nt(const float* s0) {
  v4f x0 = __builtin_nontemporal_load((const v4f*)(s0 + 0));
  v4f x1 = __builtin_nontemporal_load((const v4f*)(s0 + 4));
  v4f x2 = __builtin_nontemporal_load((const v4f*)(s0 + 16));
  v4f x3 = __builtin_nontemporal_load((const v4f*)(s0 + 20));
  v16h a;
#pragma unroll
  for (int i = 0; i < 4; ++i) {
    a[i]      = (_Float16)x0[i];
    a[i + 4]  = (_Float16)x1[i];
    a[i + 8]  = (_Float16)x2[i];
    a[i + 12] = (_Float16)x3[i];
  }
  return a;
}

__device__ __forceinline__ v8f wmma_f16(v16h a, v16h b, v8f c) {
  return __builtin_amdgcn_wmma_f32_16x16x32_f16(false, a, false, b, (short)0, c,
                                                false, false);
}

// ---------------------------------------------------------------------------
// GEMM1: h_pre = x[Nc,256] @ Wup[256,128]  (fp32 in, f16 out) + BN0 stats
// ---------------------------------------------------------------------------
#define WSTRIDE1 264  // 256 + 8 halves pad -> breaks LDS bank aliasing

__global__ __launch_bounds__(128, 4) void fvdb_gemm1(
    const float* __restrict__ x, const float* __restrict__ Wup,
    _Float16* __restrict__ h_pre, float* __restrict__ gsum,
    float* __restrict__ gsq, int Nc) {
  __shared__ _Float16 sBt[128 * WSTRIDE1];
  __shared__ float sSum[128], sSq[128];
  const int tid = threadIdx.x;
  for (int idx = tid; idx < 256 * 128; idx += 128) {
    int k = idx >> 7, n = idx & 127;
    sBt[n * WSTRIDE1 + k] = (_Float16)Wup[idx];
  }
  if (tid < 128) { sSum[tid] = 0.f; sSq[tid] = 0.f; }
  __syncthreads();

  const int lane = tid & 31, wave = tid >> 5;
  const int m = lane & 15, g = lane >> 4;
  const int numTiles = (Nc + 63) >> 6;

  for (int tile = blockIdx.x; tile < numTiles; tile += gridDim.x) {
    const int rowBase = tile * 64 + wave * 16;
    const int arow = rowBase + m;
    const float* xr = x + (size_t)(arow < Nc ? arow : 0) * 256;

    v8f acc[8] = {};
#pragma unroll 1  // keep rolled: prevents LICM of LDS B-frags -> no spills
    for (int kk = 0; kk < 8; ++kk) {
      const int k0 = kk * 32;
      v16h a = load_a_f32_nt(xr + k0 + 8 * g);
#pragma unroll
      for (int t = 0; t < 8; ++t) {
        v16h b = load_wt_frag(sBt + (t * 16 + m) * WSTRIDE1, k0, g);
        acc[t] = wmma_f16(a, b, acc[t]);
      }
    }
    if ((tile + 1) * 64 <= Nc) {  // uniform: interior tile -> scalar branch
      _Float16* pbase = h_pre + (size_t)(rowBase + 8 * g) * 128 + m;
#pragma unroll
      for (int t = 0; t < 8; ++t) {
        const int col = t * 16 + m;
        float ps = 0.f, pq = 0.f;
#pragma unroll
        for (int r = 0; r < 8; ++r) {
          float v = acc[t][r];
          pbase[t * 16 + r * 128] = (_Float16)v;  // immediate-offset stores
          ps += v;
          pq += v * v;
        }
        atomicAdd(&sSum[col], ps);
        atomicAdd(&sSq[col], pq);
      }
    } else {  // tail tile: guarded
#pragma unroll
      for (int t = 0; t < 8; ++t) {
        const int col = t * 16 + m;
        float ps = 0.f, pq = 0.f;
#pragma unroll
        for (int r = 0; r < 8; ++r) {
          const int orow = rowBase + 8 * g + r;
          float v = acc[t][r];
          if (orow < Nc) {
            h_pre[(size_t)orow * 128 + col] = (_Float16)v;
            ps += v;
            pq += v * v;
          }
        }
        atomicAdd(&sSum[col], ps);
        atomicAdd(&sSq[col], pq);
      }
    }
  }
  __syncthreads();
  if (tid < 128) {
    atomicAdd(&gsum[tid], sSum[tid]);
    atomicAdd(&gsq[tid], sSq[tid]);
  }
}

// ---------------------------------------------------------------------------
// finalize: scale = gamma * rsqrt(var+eps), shift = beta - mean*scale
// ---------------------------------------------------------------------------
__global__ void fvdb_finalize_bn(const float* __restrict__ sum,
                                 const float* __restrict__ sq,
                                 const float* __restrict__ gamma,
                                 const float* __restrict__ beta,
                                 float* __restrict__ scale,
                                 float* __restrict__ shift, float invN) {
  int c = threadIdx.x;
  float mean = sum[c] * invN;
  float var = sq[c] * invN - mean * mean;
  float sc = rsqrtf(var + EPS) * gamma[c];
  scale[c] = sc;
  shift[c] = beta[c] - mean * sc;
}

// ---------------------------------------------------------------------------
// apply BN0 + LeakyReLU: h_act = leaky(h_pre*scale + shift)  (f16 -> f16)
// ---------------------------------------------------------------------------
__global__ void fvdb_apply_bn(const _Float16* __restrict__ h_pre,
                              const float* __restrict__ scale,
                              const float* __restrict__ shift,
                              _Float16* __restrict__ h_act, int total) {
  int i = blockIdx.x * blockDim.x + threadIdx.x;
  if (i < total) {
    int c = i & 127;
    float v = (float)h_pre[i] * scale[c] + shift[c];
    v = v >= 0.f ? v : SLOPE * v;
    h_act[i] = (_Float16)v;
  }
}

// ---------------------------------------------------------------------------
// GEMM2: f1_pre = [gather(h_act) | skip] @ W1 + b1  + BN1 stats
// A rows built on the fly per-lane: K 0..127 from L2-resident h_act (f16),
// K 128..255 from skip (fp32 NT -> f16). kk loop split per source, rolled.
// ---------------------------------------------------------------------------
__global__ __launch_bounds__(128, 4) void fvdb_gemm2(
    const _Float16* __restrict__ h_act, const float* __restrict__ skip,
    const int* __restrict__ f2c, const float* __restrict__ W1,
    const float* __restrict__ b1, _Float16* __restrict__ f1_pre,
    float* __restrict__ gsum, float* __restrict__ gsq, int Nf) {
  __shared__ _Float16 sBt[128 * WSTRIDE1];
  __shared__ float sSum[128], sSq[128], sBias[128];
  const int tid = threadIdx.x;
  for (int idx = tid; idx < 256 * 128; idx += 128) {
    int k = idx >> 7, n = idx & 127;
    sBt[n * WSTRIDE1 + k] = (_Float16)W1[idx];
  }
  if (tid < 128) { sSum[tid] = 0.f; sSq[tid] = 0.f; sBias[tid] = b1[tid]; }
  __syncthreads();

  const int lane = tid & 31, wave = tid >> 5;
  const int m = lane & 15, g = lane >> 4;
  const int numTiles = (Nf + 63) >> 6;

  for (int tile = blockIdx.x; tile < numTiles; tile += gridDim.x) {
    const int rowBase = tile * 64 + wave * 16;
    const int arow = rowBase + m;
    const bool avalid = arow < Nf;
    const int parent = avalid ? f2c[arow] : 0;
    const _Float16* hr = h_act + (size_t)parent * 128;
    const float* sr = skip + (size_t)(avalid ? arow : 0) * 128;

    v8f acc[8] = {};
    // K = 0..127 : refined half (f16, L2-resident gather)
#pragma unroll 1
    for (int kk = 0; kk < 4; ++kk) {
      const int k0 = kk * 32;
      v8h lo = *(const v8h*)(hr + k0 + 8 * g);
      v8h hi = *(const v8h*)(hr + k0 + 16 + 8 * g);
      v16h a = frag_combine(lo, hi);
#pragma unroll
      for (int t = 0; t < 8; ++t) {
        v16h b = load_wt_frag(sBt + (t * 16 + m) * WSTRIDE1, k0, g);
        acc[t] = wmma_f16(a, b, acc[t]);
      }
    }
    // K = 128..255 : skip half (fp32 NT -> f16)
#pragma unroll 1
    for (int kk = 4; kk < 8; ++kk) {
      const int k0 = kk * 32;
      v16h a = load_a_f32_nt(sr + (k0 - 128) + 8 * g);
#pragma unroll
      for (int t = 0; t < 8; ++t) {
        v16h b = load_wt_frag(sBt + (t * 16 + m) * WSTRIDE1, k0, g);
        acc[t] = wmma_f16(a, b, acc[t]);
      }
    }
    if ((tile + 1) * 64 <= Nf) {
      _Float16* pbase = f1_pre + (size_t)(rowBase + 8 * g) * 128 + m;
#pragma unroll
      for (int t = 0; t < 8; ++t) {
        const int col = t * 16 + m;
        const float bias = sBias[col];
        float ps = 0.f, pq = 0.f;
#pragma unroll
        for (int r = 0; r < 8; ++r) {
          float v = acc[t][r] + bias;
          pbase[t * 16 + r * 128] = (_Float16)v;
          ps += v;
          pq += v * v;
        }
        atomicAdd(&sSum[col], ps);
        atomicAdd(&sSq[col], pq);
      }
    } else {
#pragma unroll
      for (int t = 0; t < 8; ++t) {
        const int col = t * 16 + m;
        const float bias = sBias[col];
        float ps = 0.f, pq = 0.f;
#pragma unroll
        for (int r = 0; r < 8; ++r) {
          const int orow = rowBase + 8 * g + r;
          float v = acc[t][r] + bias;
          if (orow < Nf) {
            f1_pre[(size_t)orow * 128 + col] = (_Float16)v;
            ps += v;
            pq += v * v;
          }
        }
        atomicAdd(&sSum[col], ps);
        atomicAdd(&sSq[col], pq);
      }
    }
  }
  __syncthreads();
  if (tid < 128) {
    atomicAdd(&gsum[tid], sSum[tid]);
    atomicAdd(&gsq[tid], sSq[tid]);
  }
}

// ---------------------------------------------------------------------------
// GEMM3: f2_pre = leaky(bn1(f1_pre)) @ W2 + b2  + BN2 stats
// BN1 affine + LeakyReLU fused into A-fragment construction.
// ---------------------------------------------------------------------------
#define WSTRIDE2 136  // 128 + 8 halves pad

__global__ __launch_bounds__(128, 4) void fvdb_gemm3(
    const _Float16* __restrict__ f1_pre, const float* __restrict__ W2,
    const float* __restrict__ b2, const float* __restrict__ scale1,
    const float* __restrict__ shift1, _Float16* __restrict__ f2_pre,
    float* __restrict__ gsum, float* __restrict__ gsq, int Nf) {
  __shared__ _Float16 sBt[128 * WSTRIDE2];
  __shared__ float sSum[128], sSq[128], sScale[128], sShift[128], sBias[128];
  const int tid = threadIdx.x;
  for (int idx = tid; idx < 128 * 128; idx += 128) {
    int k = idx >> 7, n = idx & 127;
    sBt[n * WSTRIDE2 + k] = (_Float16)W2[idx];
  }
  if (tid < 128) {
    sSum[tid] = 0.f; sSq[tid] = 0.f;
    sScale[tid] = scale1[tid]; sShift[tid] = shift1[tid]; sBias[tid] = b2[tid];
  }
  __syncthreads();

  const int lane = tid & 31, wave = tid >> 5;
  const int m = lane & 15, g = lane >> 4;
  const int numTiles = (Nf + 63) >> 6;

  for (int tile = blockIdx.x; tile < numTiles; tile += gridDim.x) {
    const int rowBase = tile * 64 + wave * 16;
    const int arow = rowBase + m;
    const _Float16* fr = f1_pre + (size_t)(arow < Nf ? arow : 0) * 128;

    v8f acc[8] = {};
#pragma unroll 1
    for (int kk = 0; kk < 4; ++kk) {
      const int k0 = kk * 32;
      const int kb = k0 + 8 * g;
      v16h a;
#pragma unroll
      for (int i = 0; i < 8; ++i) {
        int ka = kb + i, kc = kb + 16 + i;
        float va = (float)fr[ka] * sScale[ka] + sShift[ka];
        va = va >= 0.f ? va : SLOPE * va;
        float vc = (float)fr[kc] * sScale[kc] + sShift[kc];
        vc = vc >= 0.f ? vc : SLOPE * vc;
        a[i] = (_Float16)va;
        a[i + 8] = (_Float16)vc;
      }
#pragma unroll
      for (int t = 0; t < 8; ++t) {
        v16h b = load_wt_frag(sBt + (t * 16 + m) * WSTRIDE2, k0, g);
        acc[t] = wmma_f16(a, b, acc[t]);
      }
    }
    if ((tile + 1) * 64 <= Nf) {
      _Float16* pbase = f2_pre + (size_t)(rowBase + 8 * g) * 128 + m;
#pragma unroll
      for (int t = 0; t < 8; ++t) {
        const int col = t * 16 + m;
        const float bias = sBias[col];
        float ps = 0.f, pq = 0.f;
#pragma unroll
        for (int r = 0; r < 8; ++r) {
          float v = acc[t][r] + bias;
          pbase[t * 16 + r * 128] = (_Float16)v;
          ps += v;
          pq += v * v;
        }
        atomicAdd(&sSum[col], ps);
        atomicAdd(&sSq[col], pq);
      }
    } else {
#pragma unroll
      for (int t = 0; t < 8; ++t) {
        const int col = t * 16 + m;
        const float bias = sBias[col];
        float ps = 0.f, pq = 0.f;
#pragma unroll
        for (int r = 0; r < 8; ++r) {
          const int orow = rowBase + 8 * g + r;
          float v = acc[t][r] + bias;
          if (orow < Nf) {
            f2_pre[(size_t)orow * 128 + col] = (_Float16)v;
            ps += v;
            pq += v * v;
          }
        }
        atomicAdd(&sSum[col], ps);
        atomicAdd(&sSq[col], pq);
      }
    }
  }
  __syncthreads();
  if (tid < 128) {
    atomicAdd(&gsum[tid], sSum[tid]);
    atomicAdd(&gsq[tid], sSq[tid]);
  }
}

// ---------------------------------------------------------------------------
// final: out = leaky(bn2(f2_pre)) + refined   (refined re-gathered from the
// L2-resident h_act table instead of a 256MB materialized buffer)
// ---------------------------------------------------------------------------
__global__ void fvdb_final(const _Float16* __restrict__ f2_pre,
                           const _Float16* __restrict__ h_act,
                           const int* __restrict__ f2c,
                           const float* __restrict__ scale,
                           const float* __restrict__ shift,
                           float* __restrict__ out, int total) {
  int i = blockIdx.x * blockDim.x + threadIdx.x;
  if (i < total) {
    int row = i >> 7, c = i & 127;
    float v = (float)f2_pre[i] * scale[c] + shift[c];
    v = v >= 0.f ? v : SLOPE * v;
    float res = v + (float)h_act[(size_t)f2c[row] * 128 + c];
    __builtin_nontemporal_store(res, &out[i]);  // never re-read
  }
}

// ---------------------------------------------------------------------------
extern "C" void kernel_launch(void* const* d_in, const int* in_sizes, int n_in,
                              void* d_out, int out_size, void* d_ws,
                              size_t ws_size, hipStream_t stream) {
  (void)n_in; (void)out_size; (void)ws_size;
  const float* x    = (const float*)d_in[0];
  const float* skip = (const float*)d_in[1];
  const int*   f2c  = (const int*)d_in[2];
  const float* Wup  = (const float*)d_in[3];
  const float* upg  = (const float*)d_in[4];
  const float* upb  = (const float*)d_in[5];
  const float* W1   = (const float*)d_in[6];
  const float* b1   = (const float*)d_in[7];
  const float* g1   = (const float*)d_in[8];
  const float* bt1  = (const float*)d_in[9];
  const float* W2   = (const float*)d_in[10];
  const float* b2   = (const float*)d_in[11];
  const float* g2   = (const float*)d_in[12];
  const float* bt2  = (const float*)d_in[13];
  float* out = (float*)d_out;

  const int Nc = in_sizes[0] / 256;
  const int Nf = in_sizes[2];

  uint8_t* wsb = (uint8_t*)d_ws;
  float* stats = (float*)wsb;
  float* sum0 = stats;        float* sq0 = stats + 128;
  float* sum1 = stats + 256;  float* sq1 = stats + 384;
  float* sum2 = stats + 512;  float* sq2 = stats + 640;
  float* scale0 = stats + 768;  float* shift0 = stats + 896;
  float* scale1 = stats + 1024; float* shift1 = stats + 1152;
  float* scale2 = stats + 1280; float* shift2 = stats + 1408;

  size_t off = 8192;
  _Float16* h_pre = (_Float16*)(wsb + off);
  off += (size_t)Nc * 128 * 2; off = (off + 255) & ~(size_t)255;
  _Float16* h_act = (_Float16*)(wsb + off);
  off += (size_t)Nc * 128 * 2; off = (off + 255) & ~(size_t)255;
  _Float16* f1p = (_Float16*)(wsb + off);
  off += (size_t)Nf * 128 * 2; off = (off + 255) & ~(size_t)255;
  _Float16* f2p = (_Float16*)(wsb + off);

  // zero the six atomic stat accumulators (sum/sq x 3)
  hipMemsetAsync(wsb, 0, 6 * 128 * sizeof(float), stream);

  const int tiles1 = (Nc + 63) / 64;
  const int grid1 = tiles1 < 640 ? tiles1 : 640;
  const int tilesF = (Nf + 63) / 64;
  const int gridF = tilesF < 1280 ? tilesF : 1280;

  fvdb_gemm1<<<grid1, 128, 0, stream>>>(x, Wup, h_pre, sum0, sq0, Nc);
  fvdb_finalize_bn<<<1, 128, 0, stream>>>(sum0, sq0, upg, upb, scale0, shift0,
                                          1.0f / (float)Nc);
  const int nH = Nc * 128;
  fvdb_apply_bn<<<(nH + 255) / 256, 256, 0, stream>>>(h_pre, scale0, shift0,
                                                      h_act, nH);
  fvdb_gemm2<<<gridF, 128, 0, stream>>>(h_act, skip, f2c, W1, b1, f1p, sum1,
                                        sq1, Nf);
  fvdb_finalize_bn<<<1, 128, 0, stream>>>(sum1, sq1, g1, bt1, scale1, shift1,
                                          1.0f / (float)Nf);
  fvdb_gemm3<<<gridF, 128, 0, stream>>>(f1p, W2, b2, scale1, shift1, f2p, sum2,
                                        sq2, Nf);
  fvdb_finalize_bn<<<1, 128, 0, stream>>>(sum2, sq2, g2, bt2, scale2, shift2,
                                          1.0f / (float)Nf);
  const int nO = Nf * 128;
  fvdb_final<<<(nO + 255) / 256, 256, 0, stream>>>(f2p, h_act, f2c, scale2,
                                                   shift2, out, nO);
}
